// Explicit3DInteraction_82214263980112
// MI455X (gfx1250) — matrix operand order repaired and checked by reading, not compiled, not run on hardware
//
#include <hip/hip_runtime.h>
#include <hip/hip_bf16.h>
#include <cmath>

#define HDIM 128
#define EDF 294
#define EDP 320   // padded K for edge GEMM1 (multiple of 32)
#define CUTR 4.5f
#define PI_F 3.14159265358979323846f

typedef __attribute__((ext_vector_type(16))) _Float16 v16h;
typedef __attribute__((ext_vector_type(8)))  _Float16 v8h;
typedef __attribute__((ext_vector_type(8)))  float    v8f;

// ---- fast math (single-instruction v_rcp_f32 / v_exp_f32 / v_log_f32) ----
__device__ __forceinline__ float rcp_fast(float x)  { return __builtin_amdgcn_rcpf(x); }
__device__ __forceinline__ float sigmoid_f(float x) { return rcp_fast(1.0f + __expf(-x)); }
__device__ __forceinline__ float silu_f(float x)    { return x * sigmoid_f(x); }
__device__ __forceinline__ float softplus_f(float x){ return (x > 20.0f) ? x : __logf(1.0f + __expf(x)); }
__device__ __forceinline__ float tanh_fast(float x) {
  float e = __expf(2.0f * x);
  return (e - 1.0f) * rcp_fast(e + 1.0f);
}

// A fragment: 16x32 f16, row-major source with lda stride built into `row` ptr.
// lanes 0-15: M=lane, K = k0+{0..7} and k0+16+{0..7}
// lanes 16-31: M=lane-16, K = k0+8+{0..7} and k0+24+{0..7}
__device__ __forceinline__ v16h ldA(const _Float16* row, int k0, int hi) {
  v8h a0 = *(const v8h*)(row + k0 + hi * 8);
  v8h a1 = *(const v8h*)(row + k0 + 16 + hi * 8);
  v16h r;
#pragma unroll
  for (int i = 0; i < 8; ++i) { r[i] = a0[i]; r[i + 8] = a1[i]; }
  return r;
}

// B fragment: 32x16 f16, weights stored N-major (col ptr = Wt + n*ldk).
// lanes 0-15 hold K = k0+{0..7} and k0+16+{0..7} ; lanes 16-31 hold K = k0+8+{0..7} and k0+24+{0..7}
__device__ __forceinline__ v16h ldB(const _Float16* col, int k0, int hi) {
  v8h b0 = *(const v8h*)(col + k0 + hi * 8);
  v8h b1 = *(const v8h*)(col + k0 + 16 + hi * 8);
  v16h r;
#pragma unroll
  for (int i = 0; i < 8; ++i) { r[i] = b0[i]; r[i + 8] = b1[i]; }
  return r;
}

#define WMMA_F16(a, b, c) __builtin_amdgcn_wmma_f32_16x16x32_f16(false, (a), false, (b), (short)0, (c), false, false)

// ---------------------------------------------------------------------------
// Weight prep: transpose to f16 N-major (K contiguous per output column).
// ---------------------------------------------------------------------------
__global__ void prep_weights(const float* __restrict__ phys_w1, const float* __restrict__ ms_w,
                             const float* __restrict__ mv_w,  const float* __restrict__ em_w1,
                             const float* __restrict__ em_w2, const float* __restrict__ cs_w1,
                             const float* __restrict__ cv_w1, const float* __restrict__ us_w,
                             const float* __restrict__ uv_w,
                             _Float16* w0, _Float16* w1, _Float16* w2, _Float16* w3,
                             _Float16* w4, _Float16* w5, _Float16* w6, _Float16* w7) {
  int mat = blockIdx.y;
  int tid = blockIdx.x * blockDim.x + threadIdx.x;
  switch (mat) {
    case 0: if (tid < 128 * 128) { int n = tid / 128, k = tid % 128; w0[tid] = (_Float16)phys_w1[k * 128 + n]; } break;
    case 1: if (tid < 128 * 128) { int n = tid / 128, k = tid % 128; w1[tid] = (_Float16)ms_w[k * 128 + n]; } break;
    case 2: if (tid < 128 * 128) { int n = tid / 128, k = tid % 128; w2[tid] = (_Float16)mv_w[k * 128 + n]; } break;
    case 3: if (tid < 128 * EDP) { int n = tid / EDP, k = tid % EDP;
              w3[tid] = (k < EDF) ? (_Float16)em_w1[k * 128 + n] : (_Float16)0.0f; } break;
    case 4: if (tid < 128 * 128) { int n = tid / 128, k = tid % 128; w4[tid] = (_Float16)em_w2[k * 128 + n]; } break;
    case 5: if (tid < 128 * 128) { int n = tid / 128, k = tid % 128;
              w5[tid] = (n < 64) ? (_Float16)cs_w1[k * 64 + n] : (_Float16)cv_w1[k * 64 + (n - 64)]; } break;
    case 6: if (tid < 128 * 256) { int n = tid / 256, k = tid % 256; w6[tid] = (_Float16)us_w[k * 128 + n]; } break;
    case 7: if (tid < 128 * 256) { int n = tid / 256, k = tid % 256; w7[tid] = (_Float16)uv_w[k * 128 + n]; } break;
  }
}

__global__ void zero_kernel(float* p, size_t n) {
  size_t i = (size_t)blockIdx.x * blockDim.x + threadIdx.x;
  size_t stride = (size_t)gridDim.x * blockDim.x;
  for (; i < n; i += stride) p[i] = 0.0f;
}

// ---------------------------------------------------------------------------
// Per-node: phys MLP (softplus head) + message projection P = h@W + b
// One wave per 16-node tile; blockDim = 64 (2 waves).
// ---------------------------------------------------------------------------
__global__ __launch_bounds__(64) void node_proj(const float* __restrict__ h, int N,
                                                const _Float16* __restrict__ pw1t, const float* __restrict__ pb1,
                                                const float* __restrict__ pw2, const float* __restrict__ pb2,
                                                const _Float16* __restrict__ wmt, const float* __restrict__ mb,
                                                float* __restrict__ lj, float* __restrict__ P) {
  __shared__ __align__(16) _Float16 sA[2][16][HDIM];
  __shared__ float sT[2][16][HDIM];
  const int lane = threadIdx.x & 31, w = threadIdx.x >> 5;
  const int hi = lane >> 4, n16 = lane & 15;
  const int base = (blockIdx.x * 2 + w) * 16;

  for (int m = 0; m < 16; ++m) {
    int node = min(base + m, N - 1);
    for (int j = lane; j < HDIM; j += 32) sA[w][m][j] = (_Float16)h[(size_t)node * HDIM + j];
  }
  __syncthreads();

  const _Float16* arow = &sA[w][n16][0];
  // phys layer 1: 16x128 = A(16x128) @ W(128x128)
  for (int nt = 0; nt < 8; ++nt) {
    v8f acc = {};
    const int col = nt * 16 + n16;
    const _Float16* bcol = pw1t + (size_t)col * HDIM;
    for (int k0 = 0; k0 < HDIM; k0 += 32)
      acc = WMMA_F16(ldA(arow, k0, hi), ldB(bcol, k0, hi), acc);
    float bb = pb1[col];
#pragma unroll
    for (int r = 0; r < 8; ++r) sT[w][r + hi * 8][col] = silu_f(acc[r] + bb);
  }
  __syncthreads();

  // phys layer 2: 128 -> 2, softplus. lane -> (row n16, out hi)
  {
    float dot = 0.0f;
    for (int k = 0; k < HDIM; ++k) dot += sT[w][n16][k] * pw2[k * 2 + hi];
    dot += pb2[hi];
    if (base + n16 < N) lj[(size_t)(base + n16) * 2 + hi] = softplus_f(dot) + 1e-3f;
  }

  // message projection: P = h @ W + b
  for (int nt = 0; nt < 8; ++nt) {
    v8f acc = {};
    const int col = nt * 16 + n16;
    const _Float16* bcol = wmt + (size_t)col * HDIM;
    for (int k0 = 0; k0 < HDIM; k0 += 32)
      acc = WMMA_F16(ldA(arow, k0, hi), ldB(bcol, k0, hi), acc);
    float bb = mb[col];
#pragma unroll
    for (int r = 0; r < 8; ++r) {
      int node = base + r + hi * 8;
      if (node < N) P[(size_t)node * HDIM + col] = acc[r] + bb;
    }
  }
}

// ---------------------------------------------------------------------------
// Edge kernel: features + LN + edge MLP (2 WMMA GEMMs) + gate/scatter +
// combined coord heads (1 WMMA GEMM). One wave per 16-edge tile, blockDim 64.
// ---------------------------------------------------------------------------
__global__ __launch_bounds__(64) void edge_kernel(
    const float* __restrict__ h_s, const float* __restrict__ h_v,
    const float* __restrict__ p_s, const float* __restrict__ p_v,
    const float* __restrict__ q_s, const float* __restrict__ q_v,
    const int* __restrict__ eidx, int E,
    const float* __restrict__ lj_s, const float* __restrict__ lj_v,
    const float* __restrict__ en_g, const float* __restrict__ en_b,
    const _Float16* __restrict__ w1t, const float* __restrict__ em_b1,
    const _Float16* __restrict__ w2t, const float* __restrict__ em_b2,
    const float* __restrict__ pg_w, const float* __restrict__ pg_b,
    const float* __restrict__ Ps, const float* __restrict__ Pv,
    const _Float16* __restrict__ c1t,
    const float* __restrict__ cs_b1, const float* __restrict__ cv_b1,
    const float* __restrict__ cs_w2, const float* __restrict__ cs_b2,
    const float* __restrict__ cv_w2, const float* __restrict__ cv_b2,
    float* __restrict__ sum_s, float* __restrict__ sum_v,
    float* __restrict__ cnt_s, float* __restrict__ cnt_v,
    float* __restrict__ dsum_s, float* __restrict__ dsum_v) {
  __shared__ __align__(16) _Float16 sFeat[2][16][EDP];
  __shared__ __align__(16) _Float16 sEh1[2][16][HDIM];
  __shared__ __align__(16) _Float16 sEh[2][16][HDIM];
  __shared__ float sS[2][16][10];
  __shared__ int sIs[2][16], sIv[2][16];

  const int lane = threadIdx.x & 31, w = threadIdx.x >> 5;
  const int hi = lane >> 4, n16 = lane & 15;
  const int eBase = (blockIdx.x * 2 + w) * 16;

  // --- per-edge geometry / physics scalars (lanes 0..15) ---
  if (lane < 16) {
    int e = min(eBase + lane, E - 1);
    bool valid = (eBase + lane) < E;
    int is = eidx[e], iv = eidx[E + e];
    sIs[w][lane] = is; sIv[w][lane] = iv;
    // prefetch the gather rows used by the feature build + message scatter
    const char* phs = (const char*)(h_s + (size_t)is * HDIM);
    const char* phv = (const char*)(h_v + (size_t)iv * HDIM);
    const char* pps = (const char*)(Ps + (size_t)is * HDIM);
    const char* ppv = (const char*)(Pv + (size_t)iv * HDIM);
    __builtin_prefetch(phs, 0, 3); __builtin_prefetch(phs + 256, 0, 3);
    __builtin_prefetch(phv, 0, 3); __builtin_prefetch(phv + 256, 0, 3);
    __builtin_prefetch(pps, 0, 3); __builtin_prefetch(pps + 256, 0, 3);
    __builtin_prefetch(ppv, 0, 3); __builtin_prefetch(ppv + 256, 0, 3);
    float rx = p_s[is * 3 + 0] - p_v[iv * 3 + 0];
    float ry = p_s[is * 3 + 1] - p_v[iv * 3 + 1];
    float rz = p_s[is * 3 + 2] - p_v[iv * 3 + 2];
    float dist = sqrtf(rx * rx + ry * ry + rz * rz + 1e-8f);
    float safe = fmaxf(dist, 0.5f);
    float inv = rcp_fast(safe);
    float eps = sqrtf(lj_s[is * 2 + 0] * lj_v[iv * 2 + 0] + 1e-8f);
    float sig = 0.5f * (lj_s[is * 2 + 1] + lj_v[iv * 2 + 1]);
    float ratio = fminf(sig * inv, 1.5f);
    float r3 = ratio * ratio * ratio, r6 = r3 * r3, r12 = r6 * r6;
    float elj = 4.0f * eps * (r12 - r6);
    float qs = q_s[is], qv = q_v[iv];
    float ecoul = qs * qv * inv;
    sS[w][lane][0] = dist;      sS[w][lane][1] = inv;
    sS[w][lane][2] = rx * inv;  sS[w][lane][3] = ry * inv;  sS[w][lane][4] = rz * inv;
    sS[w][lane][5] = elj;       sS[w][lane][6] = ecoul;
    sS[w][lane][7] = qs * qv;   sS[w][lane][8] = fabsf(qs - qv);
    sS[w][lane][9] = valid ? 1.0f : 0.0f;
    if (valid) { atomicAdd(&cnt_s[is], 1.0f); atomicAdd(&cnt_v[iv], 1.0f); }
  }
  __syncthreads();

  // --- build normalized edge features in LDS (f16) ---
  for (int e = 0; e < 16; ++e) {
    int is = sIs[w][e], iv = sIv[w][e];
    float dist = sS[w][e][0], inv = sS[w][e][1];
    float elj = sS[w][e][5], ecoul = sS[w][e][6], qq = sS[w][e][7], adq = sS[w][e][8];
    float clip = fmaxf(dist, 1e-6f);
    float rclip = rcp_fast(clip);
    float s = 0.0f, sq = 0.0f;
    for (int j = lane; j < EDP; j += 32) {
      float f = 0.0f;
      if (j < 128)       f = h_s[(size_t)is * HDIM + j];
      else if (j < 256)  f = h_v[(size_t)iv * HDIM + (j - 128)];
      else if (j < 288)  { float fr = (float)(j - 255) * (PI_F / CUTR); f = __sinf(fr * clip) * rclip; }
      else if (j == 288) f = dist * (1.0f / CUTR);
      else if (j == 289) f = inv;
      else if (j == 290) f = qq;
      else if (j == 291) f = adq;
      else if (j == 292) f = elj;
      else if (j == 293) f = ecoul;
      sFeat[w][e][j] = (_Float16)f;
      if (j < EDF) { s += f; sq += f * f; }
    }
    for (int off = 1; off < 32; off <<= 1) { s += __shfl_xor(s, off, 32); sq += __shfl_xor(sq, off, 32); }
    float mean = s * (1.0f / (float)EDF);
    float var = sq * (1.0f / (float)EDF) - mean * mean;
    float rstd = rsqrtf(var + 1e-5f);
    for (int j = lane; j < EDF; j += 32) {
      float f = (float)sFeat[w][e][j];
      sFeat[w][e][j] = (_Float16)((f - mean) * rstd * en_g[j] + en_b[j]);
    }
  }
  __syncthreads();

  // --- GEMM1: eh1 = silu(feat(16x320) @ W1 + b1) ---
  {
    const _Float16* arow = &sFeat[w][n16][0];
    for (int nt = 0; nt < 8; ++nt) {
      v8f acc = {};
      const int col = nt * 16 + n16;
      const _Float16* bcol = w1t + (size_t)col * EDP;
      for (int k0 = 0; k0 < EDP; k0 += 32)
        acc = WMMA_F16(ldA(arow, k0, hi), ldB(bcol, k0, hi), acc);
      float bb = em_b1[col];
#pragma unroll
      for (int r = 0; r < 8; ++r) sEh1[w][r + hi * 8][col] = (_Float16)silu_f(acc[r] + bb);
    }
  }
  __syncthreads();

  // --- GEMM2: eh = silu(eh1 @ W2 + b2); gate + message scatter ---
  {
    const _Float16* arow = &sEh1[w][n16][0];
    for (int nt = 0; nt < 8; ++nt) {
      v8f acc = {};
      const int col = nt * 16 + n16;
      const _Float16* bcol = w2t + (size_t)col * HDIM;
      for (int k0 = 0; k0 < HDIM; k0 += 32)
        acc = WMMA_F16(ldA(arow, k0, hi), ldB(bcol, k0, hi), acc);
      float bb = em_b2[col];
      float pg0 = pg_w[col], pg1 = pg_w[HDIM + col], pgb = pg_b[col];
#pragma unroll
      for (int r = 0; r < 8; ++r) {
        int m = r + hi * 8;
        float ehv = silu_f(acc[r] + bb);
        sEh[w][m][col] = (_Float16)ehv;
        if (sS[w][m][9] > 0.0f) {
          float g = sigmoid_f(ehv) *
                    (sigmoid_f(sS[w][m][5] * pg0 + sS[w][m][6] * pg1 + pgb) + 0.05f);
          int is = sIs[w][m], iv = sIv[w][m];
          atomicAdd(&sum_s[(size_t)is * HDIM + col], Pv[(size_t)iv * HDIM + col] * g);
          atomicAdd(&sum_v[(size_t)iv * HDIM + col], Ps[(size_t)is * HDIM + col] * g);
        }
      }
    }
  }
  __syncthreads();

  // --- GEMM3: combined coord heads. cols 0..63 -> w_s path, 64..127 -> w_v ---
  {
    const _Float16* arow = &sEh[w][n16][0];
    float pws[8], pwv[8];
#pragma unroll
    for (int r = 0; r < 8; ++r) { pws[r] = 0.0f; pwv[r] = 0.0f; }
    for (int nt = 0; nt < 8; ++nt) {
      v8f acc = {};
      const int col = nt * 16 + n16;
      const _Float16* bcol = c1t + (size_t)col * HDIM;
      for (int k0 = 0; k0 < HDIM; k0 += 32)
        acc = WMMA_F16(ldA(arow, k0, hi), ldB(bcol, k0, hi), acc);
      bool sPath = (col < 64);
      float b1 = sPath ? cs_b1[col] : cv_b1[col - 64];
      float w2 = sPath ? cs_w2[col] : cv_w2[col - 64];
#pragma unroll
      for (int r = 0; r < 8; ++r) {
        float t = silu_f(acc[r] + b1) * w2;
        if (sPath) pws[r] += t; else pwv[r] += t;
      }
    }
#pragma unroll
    for (int r = 0; r < 8; ++r) {
      float a = pws[r], b = pwv[r];
      for (int off = 1; off < 16; off <<= 1) { a += __shfl_xor(a, off, 32); b += __shfl_xor(b, off, 32); }
      int m = r + hi * 8;
      if (n16 == 0 && sS[w][m][9] > 0.0f) {
        float wsv = 0.12f * tanh_fast(a + cs_b2[0]);
        float wvv = 0.12f * tanh_fast(b + cv_b2[0]);
        int is = sIs[w][m], iv = sIv[w][m];
        float ux = sS[w][m][2], uy = sS[w][m][3], uz = sS[w][m][4];
        atomicAdd(&dsum_s[(size_t)is * 3 + 0], ux * wsv);
        atomicAdd(&dsum_s[(size_t)is * 3 + 1], uy * wsv);
        atomicAdd(&dsum_s[(size_t)is * 3 + 2], uz * wsv);
        atomicAdd(&dsum_v[(size_t)iv * 3 + 0], -ux * wvv);
        atomicAdd(&dsum_v[(size_t)iv * 3 + 1], -uy * wvv);
        atomicAdd(&dsum_v[(size_t)iv * 3 + 2], -uz * wvv);
      }
    }
  }
}

// ---------------------------------------------------------------------------
// Node update: out = h + LN(silu(concat[h, agg] @ W + b))
// ---------------------------------------------------------------------------
__global__ __launch_bounds__(64) void node_update(const float* __restrict__ h,
                                                  const float* __restrict__ sum,
                                                  const float* __restrict__ cnt, int N,
                                                  const _Float16* __restrict__ wt,
                                                  const float* __restrict__ b,
                                                  const float* __restrict__ g,
                                                  const float* __restrict__ beta,
                                                  float* __restrict__ out) {
  __shared__ __align__(16) _Float16 sA[2][16][256];
  __shared__ float sO[2][16][HDIM];
  const int lane = threadIdx.x & 31, w = threadIdx.x >> 5;
  const int hi = lane >> 4, n16 = lane & 15;
  const int base = (blockIdx.x * 2 + w) * 16;

  for (int m = 0; m < 16; ++m) {
    int node = min(base + m, N - 1);
    float ic = rcp_fast(fmaxf(cnt[node], 1.0f));
    for (int j = lane; j < 256; j += 32)
      sA[w][m][j] = (j < HDIM) ? (_Float16)h[(size_t)node * HDIM + j]
                               : (_Float16)(sum[(size_t)node * HDIM + (j - HDIM)] * ic);
  }
  __syncthreads();

  const _Float16* arow = &sA[w][n16][0];
  for (int nt = 0; nt < 8; ++nt) {
    v8f acc = {};
    const int col = nt * 16 + n16;
    const _Float16* bcol = wt + (size_t)col * 256;
    for (int k0 = 0; k0 < 256; k0 += 32)
      acc = WMMA_F16(ldA(arow, k0, hi), ldB(bcol, k0, hi), acc);
    float bb = b[col];
#pragma unroll
    for (int r = 0; r < 8; ++r) sO[w][r + hi * 8][col] = silu_f(acc[r] + bb);
  }
  __syncthreads();

  // row LayerNorm + residual: row = n16, half = hi
  float s = 0.0f, sq = 0.0f;
  for (int c = 0; c < 64; ++c) { float x = sO[w][n16][hi * 64 + c]; s += x; sq += x * x; }
  s += __shfl_xor(s, 16, 32); sq += __shfl_xor(sq, 16, 32);
  float mean = s * (1.0f / HDIM);
  float var = sq * (1.0f / HDIM) - mean * mean;
  float rstd = rsqrtf(var + 1e-5f);
  int node = base + n16;
  if (node < N) {
    for (int c = 0; c < 64; ++c) {
      int col = hi * 64 + c;
      float x = sO[w][n16][col];
      out[(size_t)node * HDIM + col] =
          h[(size_t)node * HDIM + col] + (x - mean) * rstd * g[col] + beta[col];
    }
  }
}

__global__ void pos_update(const float* __restrict__ p, const float* __restrict__ dsum,
                           const float* __restrict__ cnt, float* __restrict__ out, int N) {
  int i = blockIdx.x * blockDim.x + threadIdx.x;
  if (i < N) {
    float ic = rcp_fast(fmaxf(cnt[i], 1.0f));
    out[(size_t)i * 3 + 0] = p[(size_t)i * 3 + 0] + dsum[(size_t)i * 3 + 0] * ic;
    out[(size_t)i * 3 + 1] = p[(size_t)i * 3 + 1] + dsum[(size_t)i * 3 + 1] * ic;
    out[(size_t)i * 3 + 2] = p[(size_t)i * 3 + 2] + dsum[(size_t)i * 3 + 2] * ic;
  }
}

// ---------------------------------------------------------------------------
extern "C" void kernel_launch(void* const* d_in, const int* in_sizes, int n_in,
                              void* d_out, int out_size, void* d_ws, size_t ws_size,
                              hipStream_t stream) {
  const float* h_s = (const float*)d_in[0];
  const float* p_s = (const float*)d_in[1];
  const float* h_v = (const float*)d_in[3];
  const float* p_v = (const float*)d_in[4];
  const float* q_s = (const float*)d_in[6];
  const float* q_v = (const float*)d_in[7];
  const int*   eix = (const int*)d_in[8];
  const float* phys_w1 = (const float*)d_in[9];
  const float* phys_b1 = (const float*)d_in[10];
  const float* phys_w2 = (const float*)d_in[11];
  const float* phys_b2 = (const float*)d_in[12];
  const float* en_g = (const float*)d_in[13];
  const float* en_b = (const float*)d_in[14];
  const float* em_w1 = (const float*)d_in[15];
  const float* em_b1 = (const float*)d_in[16];
  const float* em_w2 = (const float*)d_in[17];
  const float* em_b2 = (const float*)d_in[18];
  const float* pg_w = (const float*)d_in[19];
  const float* pg_b = (const float*)d_in[20];
  const float* ms_w = (const float*)d_in[21];
  const float* ms_b = (const float*)d_in[22];
  const float* mv_w = (const float*)d_in[23];
  const float* mv_b = (const float*)d_in[24];
  const float* us_w = (const float*)d_in[25];
  const float* us_b = (const float*)d_in[26];
  const float* us_g = (const float*)d_in[27];
  const float* us_beta = (const float*)d_in[28];
  const float* uv_w = (const float*)d_in[29];
  const float* uv_b = (const float*)d_in[30];
  const float* uv_g = (const float*)d_in[31];
  const float* uv_beta = (const float*)d_in[32];
  const float* cs_w1 = (const float*)d_in[33];
  const float* cs_b1 = (const float*)d_in[34];
  const float* cs_w2 = (const float*)d_in[35];
  const float* cs_b2 = (const float*)d_in[36];
  const float* cv_w1 = (const float*)d_in[37];
  const float* cv_b1 = (const float*)d_in[38];
  const float* cv_w2 = (const float*)d_in[39];
  const float* cv_b2 = (const float*)d_in[40];

  const int NS = in_sizes[0] / HDIM;
  const int NV = in_sizes[3] / HDIM;
  const int E  = in_sizes[8] / 2;

  char* base = (char*)d_ws;
  size_t off = 0;
  auto alloc = [&](size_t elems, size_t esz) -> void* {
    void* p = base + off;
    off = (off + elems * esz + 255) & ~(size_t)255;
    return p;
  };

  _Float16* w_phys1 = (_Float16*)alloc((size_t)128 * 128, 2);
  _Float16* w_ms    = (_Float16*)alloc((size_t)128 * 128, 2);
  _Float16* w_mv    = (_Float16*)alloc((size_t)128 * 128, 2);
  _Float16* w_em1   = (_Float16*)alloc((size_t)128 * EDP, 2);
  _Float16* w_em2   = (_Float16*)alloc((size_t)128 * 128, 2);
  _Float16* w_c1    = (_Float16*)alloc((size_t)128 * 128, 2);
  _Float16* w_us    = (_Float16*)alloc((size_t)128 * 256, 2);
  _Float16* w_uv    = (_Float16*)alloc((size_t)128 * 256, 2);

  float* lj_s = (float*)alloc((size_t)NS * 2, 4);
  float* lj_v = (float*)alloc((size_t)NV * 2, 4);
  float* Ps   = (float*)alloc((size_t)NS * HDIM, 4);
  float* Pv   = (float*)alloc((size_t)NV * HDIM, 4);

  size_t accElems = (size_t)NS * HDIM + (size_t)NV * HDIM + NS + NV + (size_t)NS * 3 + (size_t)NV * 3;
  float* accBase = (float*)alloc(accElems, 4);
  float* sum_s = accBase;
  float* sum_v = sum_s + (size_t)NS * HDIM;
  float* cnt_s = sum_v + (size_t)NV * HDIM;
  float* cnt_v = cnt_s + NS;
  float* dsum_s = cnt_v + NV;
  float* dsum_v = dsum_s + (size_t)NS * 3;

  float* out = (float*)d_out;
  float* o_hs = out;
  float* o_ps = o_hs + (size_t)NS * HDIM;
  float* o_hv = o_ps + (size_t)NS * 3;
  float* o_pv = o_hv + (size_t)NV * HDIM;

  prep_weights<<<dim3(160, 8), 256, 0, stream>>>(phys_w1, ms_w, mv_w, em_w1, em_w2, cs_w1, cv_w1,
                                                 us_w, uv_w, w_phys1, w_ms, w_mv, w_em1, w_em2,
                                                 w_c1, w_us, w_uv);
  zero_kernel<<<2048, 256, 0, stream>>>(accBase, accElems);

  node_proj<<<(NS + 31) / 32, 64, 0, stream>>>(h_s, NS, w_phys1, phys_b1, phys_w2, phys_b2,
                                               w_ms, ms_b, lj_s, Ps);
  node_proj<<<(NV + 31) / 32, 64, 0, stream>>>(h_v, NV, w_phys1, phys_b1, phys_w2, phys_b2,
                                               w_mv, mv_b, lj_v, Pv);

  edge_kernel<<<(E + 31) / 32, 64, 0, stream>>>(h_s, h_v, p_s, p_v, q_s, q_v, eix, E,
                                                lj_s, lj_v, en_g, en_b, w_em1, em_b1, w_em2, em_b2,
                                                pg_w, pg_b, Ps, Pv, w_c1, cs_b1, cv_b1,
                                                cs_w2, cs_b2, cv_w2, cv_b2,
                                                sum_s, sum_v, cnt_s, cnt_v, dsum_s, dsum_v);

  node_update<<<(NS + 31) / 32, 64, 0, stream>>>(h_s, sum_s, cnt_s, NS, w_us, us_b, us_g, us_beta, o_hs);
  node_update<<<(NV + 31) / 32, 64, 0, stream>>>(h_v, sum_v, cnt_v, NV, w_uv, uv_b, uv_g, uv_beta, o_hv);

  pos_update<<<(NS + 255) / 256, 256, 0, stream>>>(p_s, dsum_s, cnt_s, o_ps, NS);
  pos_update<<<(NV + 255) / 256, 256, 0, stream>>>(p_v, dsum_v, cnt_v, o_pv, NV);
}
